// Graph_Generator_39805756900016
// MI455X (gfx1250) — compile-verified, matrix operand order checked by reading, not run on hardware
//
#include <hip/hip_runtime.h>
#include <hip/hip_bf16.h>

// SuperGlue-style matcher for MI455X (gfx1250, wave32, WMMA).
// Dominant cost (~155 GFLOP) is the two scorenets over 512x512 pixel pairs:
// implemented as f16 WMMA (v_wmma_f32_16x16x32_f16) with eval-mode batchnorm
// folded into pre-scaled f16 weights staged in (dynamic, ~85KB) LDS.
// conv1 (pairwise rank-1 part) is kept entirely in packed f16 (v_pk_add_f16 /
// v_pk_max_num_f16). WMMA accumulators start at inline C=0 (no bias preload /
// no spills); BN bias is added in the tail fused with relu+cvt. Next tile's
// B' columns are prefetched (global_prefetch_b8) to hide L2 latency.
// Transformer (~0.5 GFLOP) kept in exact f32 scalar kernels.
// Sinkhorn: 200 small logsumexp kernels in-place on the P half of d_out.
// Workspace requirement: ~5.5 MB of d_ws.

#define SG_EPS 1e-5f

typedef _Float16 v16h __attribute__((ext_vector_type(16)));
typedef _Float16 v8h  __attribute__((ext_vector_type(8)));
typedef float    v8f  __attribute__((ext_vector_type(8)));

__device__ __forceinline__ v8f wmma16(v16h a, v16h b, v8f c) {
  // D = A(16x32 f16) * B(32x16 f16) + C(16x16 f32)
  return __builtin_amdgcn_wmma_f32_16x16x32_f16(false, a, false, b, (short)0, c, false, false);
}

// ---------------- layer norm helper (blockDim.x == 64, one feature/thread) ----
__device__ __forceinline__ float ln64(float x, float g, float b, float* red, int t) {
  red[t] = x; __syncthreads();
  #pragma unroll
  for (int s = 32; s > 0; s >>= 1) { if (t < s) red[t] += red[t + s]; __syncthreads(); }
  float mean = red[0] * (1.0f / 64.0f); __syncthreads();
  float d = x - mean; red[t] = d * d; __syncthreads();
  #pragma unroll
  for (int s = 32; s > 0; s >>= 1) { if (t < s) red[t] += red[t + s]; __syncthreads(); }
  float var = red[0] * (1.0f / 64.0f); __syncthreads();
  return d * rsqrtf(var + SG_EPS) * g + b;
}

// ---------------- K1: gather descriptors + coords, init MLP ------------------
__global__ __launch_bounds__(64) void k_gather_init(
    const float* __restrict__ desc, const int* __restrict__ graph,
    const float* w1, const float* b1, const float* g1, const float* be1,
    const float* w2, const float* b2, const float* g2, const float* be2,
    const float* w3, const float* b3, float* __restrict__ X) {
  int tok = blockIdx.x; int b = tok >> 9; int t = threadIdx.x;
  __shared__ float in[66]; __shared__ float xn[64]; __shared__ float red[64];
  int gy = graph[(size_t)tok * 2 + 0], gx = graph[(size_t)tok * 2 + 1];
  int cy = gy < 0 ? 0 : (gy > 255 ? 255 : gy);
  int cx = gx < 0 ? 0 : (gx > 255 ? 255 : gx);
  in[t] = desc[(((size_t)b * 64 + t) * 256 + cy) * 256 + cx];
  if (t == 0) { in[64] = (float)gy * (1.f / 256.f); in[65] = (float)gx * (1.f / 256.f); }
  __syncthreads();
  float a = b1[t];
  #pragma unroll 6
  for (int c = 0; c < 66; c++) a += w1[t * 66 + c] * in[c];
  a = fmaxf(ln64(a, g1[t], be1[t], red, t), 0.f);
  xn[t] = a; __syncthreads();
  float a2 = b2[t];
  #pragma unroll 8
  for (int c = 0; c < 64; c++) a2 += w2[t * 64 + c] * xn[c];
  a2 = fmaxf(ln64(a2, g2[t], be2[t], red, t), 0.f);
  __syncthreads(); xn[t] = a2; __syncthreads();
  float a3 = b3[t];
  #pragma unroll 8
  for (int c = 0; c < 64; c++) a3 += w3[t * 64 + c] * xn[c];
  X[(size_t)tok * 64 + t] = a3;
}

// ---------------- K2a: QKV projections ---------------------------------------
__global__ __launch_bounds__(64) void k_qkv(
    const float* __restrict__ X,
    const float* wq, const float* bq, const float* wk, const float* bk,
    const float* wv, const float* bv,
    float* __restrict__ Q, float* __restrict__ K, float* __restrict__ V) {
  int tok = blockIdx.x; int t = threadIdx.x;
  __shared__ float xs[64];
  xs[t] = X[(size_t)tok * 64 + t]; __syncthreads();
  float q = bq[t], k = bk[t], v = bv[t];
  #pragma unroll 8
  for (int c = 0; c < 64; c++) {
    float x = xs[c];
    q += wq[t * 64 + c] * x; k += wk[t * 64 + c] * x; v += wv[t * 64 + c] * x;
  }
  Q[(size_t)tok * 64 + t] = q; K[(size_t)tok * 64 + t] = k; V[(size_t)tok * 64 + t] = v;
}

// ---------------- K2b: attention (streaming softmax, dh=8) -------------------
__global__ __launch_bounds__(256) void k_attn(
    const float* __restrict__ Q, const float* __restrict__ K,
    const float* __restrict__ V, float* __restrict__ ATT) {
  int b = blockIdx.x >> 3, h = blockIdx.x & 7; int tid = threadIdx.x;
  __shared__ float Kh[4096], Vh[4096];
  for (int idx = tid; idx < 4096; idx += 256) {
    int n = idx >> 3, d = idx & 7;
    Kh[idx] = K[((size_t)b * 512 + n) * 64 + h * 8 + d];
    Vh[idx] = V[((size_t)b * 512 + n) * 64 + h * 8 + d];
  }
  __syncthreads();
  #pragma unroll
  for (int r = 0; r < 2; r++) {
    int n = tid + r * 256;
    float q[8];
    #pragma unroll
    for (int d = 0; d < 8; d++) q[d] = Q[((size_t)b * 512 + n) * 64 + h * 8 + d];
    float m = -1e30f, s = 0.f, o[8] = {0, 0, 0, 0, 0, 0, 0, 0};
    for (int kk = 0; kk < 512; kk++) {
      float l = 0.f;
      #pragma unroll
      for (int d = 0; d < 8; d++) l += q[d] * Kh[kk * 8 + d];
      l *= 0.35355339059327f;  // 8^-0.5
      float nm = fmaxf(m, l);
      float cf = expf(m - nm);
      float e  = expf(l - nm);
      s = s * cf + e;
      #pragma unroll
      for (int d = 0; d < 8; d++) o[d] = o[d] * cf + e * Vh[kk * 8 + d];
      m = nm;
    }
    float inv = 1.f / s;
    #pragma unroll
    for (int d = 0; d < 8; d++) ATT[((size_t)b * 512 + n) * 64 + h * 8 + d] = o[d] * inv;
  }
}

// ---------------- K2c: output proj + residual + LN1 --------------------------
__global__ __launch_bounds__(64) void k_oproj(
    float* __restrict__ X, const float* __restrict__ ATT,
    const float* wo, const float* bo, const float* g, const float* be) {
  int tok = blockIdx.x; int t = threadIdx.x;
  __shared__ float as[64]; __shared__ float red[64];
  as[t] = ATT[(size_t)tok * 64 + t]; __syncthreads();
  float y = bo[t];
  #pragma unroll 8
  for (int c = 0; c < 64; c++) y += wo[t * 64 + c] * as[c];
  float xi = X[(size_t)tok * 64 + t] + y;
  X[(size_t)tok * 64 + t] = ln64(xi, g[t], be[t], red, t);
}

// ---------------- K2d: FF (64->256->64) + residual + LN2 ---------------------
__global__ __launch_bounds__(64) void k_ff(
    float* __restrict__ X,
    const float* fw1, const float* fb1, const float* fw2, const float* fb2,
    const float* g, const float* be) {
  int tok = blockIdx.x; int t = threadIdx.x;
  __shared__ float xs[64]; __shared__ float hid[256]; __shared__ float red[64];
  xs[t] = X[(size_t)tok * 64 + t]; __syncthreads();
  #pragma unroll
  for (int p = 0; p < 4; p++) {
    int k = t + 64 * p;
    float a = fb1[k];
    #pragma unroll 8
    for (int c = 0; c < 64; c++) a += fw1[k * 64 + c] * xs[c];
    hid[k] = fmaxf(a, 0.f);
  }
  __syncthreads();
  float y = fb2[t];
  #pragma unroll 8
  for (int k = 0; k < 256; k++) y += fw2[t * 256 + k] * hid[k];
  float xi = xs[t] + y;
  X[(size_t)tok * 64 + t] = ln64(xi, g[t], be[t], red, t);
}

// ---------------- K3: final LN + desc MLP ------------------------------------
__global__ __launch_bounds__(64) void k_final(
    float* __restrict__ X, const float* ng, const float* nb,
    const float* w1, const float* b1, const float* g1, const float* be1,
    const float* w2, const float* b2, const float* g2, const float* be2) {
  int tok = blockIdx.x; int t = threadIdx.x;
  __shared__ float xs[64]; __shared__ float red[64];
  float x = X[(size_t)tok * 64 + t];
  float xn0 = ln64(x, ng[t], nb[t], red, t);
  xs[t] = xn0; __syncthreads();
  float a = b1[t];
  #pragma unroll 8
  for (int c = 0; c < 64; c++) a += w1[t * 64 + c] * xs[c];
  a = fmaxf(ln64(a, g1[t], be1[t], red, t), 0.f);
  __syncthreads(); xs[t] = a; __syncthreads();
  float a2 = b2[t];
  #pragma unroll 8
  for (int c = 0; c < 64; c++) a2 += w2[t * 64 + c] * xs[c];
  a2 = fmaxf(ln64(a2, g2[t], be2[t], red, t), 0.f);
  X[(size_t)tok * 64 + t] = a2;
}

// ---------------- K4: conv1 split-rank precompute with BN1 folded (f16 out) --
// A'[c,i] = s1[c] * (W1a@x_i) ;  B'[c,j] = s1[c]*(W1b@x_j + b1 - m1) + beta1
__global__ __launch_bounds__(256) void k_ab(
    const float* __restrict__ X, const float* c1w, const float* c1b,
    const float* g1, const float* be1, const float* m1, const float* v1,
    _Float16* __restrict__ Aph, _Float16* __restrict__ Bph, int net) {
  int bx = blockIdx.x; int b = bx >> 9; int i = bx & 511; int o = threadIdx.x;
  __shared__ float xs[64];
  if (o < 64) xs[o] = X[((size_t)b * 512 + i) * 64 + o];
  __syncthreads();
  const float* w = c1w + o * 128;
  float sA = 0.f, sB = 0.f;
  #pragma unroll 8
  for (int c = 0; c < 64; c++) { sA += w[c] * xs[c]; sB += w[64 + c] * xs[c]; }
  float sc = g1[o] * rsqrtf(v1[o] + SG_EPS);
  size_t base = ((size_t)(net * 2 + b) * 512 + i) * 256 + o;
  Aph[base] = (_Float16)(sA * sc);
  Bph[base] = (_Float16)((sB + c1b[o] - m1[o]) * sc + be1[o]);
}

// ---------------- K0: scorenet weight prep (BN-fold + f16 fragment swizzle) --
// A-fragment (16x32 f16) per-lane layout: lane l owns row M = 16*mt + (l%16),
// halves h: K = 32*kt + (h<8 ? h : h+8) + (l<16 ? 0 : 8).
__global__ void k_prep(
    const float* c2w, const float* c2b, const float* g2, const float* be2,
    const float* m2, const float* v2,
    const float* c3w, const float* c3b, const float* g3, const float* be3,
    const float* m3, const float* v3,
    const float* c4w, const float* c4b,
    _Float16* __restrict__ W2H, _Float16* __restrict__ W3H,
    float* __restrict__ sb, int net) {
  int gid = blockIdx.x * blockDim.x + threadIdx.x;
  int stride = gridDim.x * blockDim.x;
  for (int idx = gid; idx < 32768; idx += stride) {
    int h = idx & 15, lane = (idx >> 4) & 31, kt = (idx >> 9) & 7, mt = idx >> 12;
    int M = 16 * mt + (lane & 15);
    int Kc = 32 * kt + (h < 8 ? h : h + 8) + ((lane < 16) ? 0 : 8);
    float sc = g2[M] * rsqrtf(v2[M] + SG_EPS);
    W2H[net * 32768 + idx] = (_Float16)(c2w[M * 256 + Kc] * sc);
  }
  for (int idx = gid; idx < 8192; idx += stride) {
    int h = idx & 15, lane = (idx >> 4) & 31, kt = (idx >> 9) & 3, mt = idx >> 11;
    int M = 16 * mt + (lane & 15);
    int Kc = 32 * kt + (h < 8 ? h : h + 8) + ((lane < 16) ? 0 : 8);
    float sc = g3[M] * rsqrtf(v3[M] + SG_EPS);
    W3H[net * 8192 + idx] = (_Float16)(c3w[M * 128 + Kc] * sc);
  }
  if (gid < 128) {
    float sc = g2[gid] * rsqrtf(v2[gid] + SG_EPS);
    sb[net * 128 + gid] = (c2b[gid] - m2[gid]) * sc + be2[gid];
  }
  if (gid < 64) {
    float sc = g3[gid] * rsqrtf(v3[gid] + SG_EPS);
    sb[256 + net * 64 + gid] = (c3b[gid] - m3[gid]) * sc + be3[gid];
    sb[384 + net * 64 + gid] = c4w[gid];
  }
  if (gid == 0) sb[512 + net] = c4b[0];
}

// ---------------- K5: scorenet main (WMMA) -----------------------------------
// grid.x = net(2) * b(2) * igroup(64) * jhalf(2) = 512 blocks of 8 waves.
// Wave owns row i; per iteration: 32 pixels (two 16-col subtiles),
// conv1 in packed f16, conv2 = 128 WMMAs (C starts at inline 0, bias in tail),
// conv3 = 32 WMMAs, conv4 = per-lane dot + shfl_xor(16).
#define SCORE_DYNLDS 87104
__global__ __launch_bounds__(256) void k_score(
    const _Float16* __restrict__ Aph, const _Float16* __restrict__ Bph,
    const _Float16* __restrict__ W2H, const _Float16* __restrict__ W3H,
    const float* __restrict__ sb,
    float* __restrict__ s1out, float* __restrict__ s2out) {
  extern __shared__ __align__(32) char smem[];
  _Float16* w2f   = (_Float16*)smem;             // 32768 halves (64KB) fragment-order
  _Float16* w3f   = (_Float16*)(smem + 65536);   // 8192 halves (16KB)
  _Float16* acolh = (_Float16*)(smem + 81920);   // 8 waves x 256 f16 A' columns (4KB)
  float* b2l      = (float*)(smem + 86016);      // 128 (bn2-folded bias)
  float* b3l      = (float*)(smem + 86528);      // 64
  float* w4l      = (float*)(smem + 86784);      // 64
  float* b4l      = (float*)(smem + 87040);      // 1

  int bx = blockIdx.x;
  int jh  = bx & 1;
  int ig  = (bx >> 1) & 63;
  int b   = (bx >> 7) & 1;
  int net = (bx >> 8) & 1;
  int tid = threadIdx.x;
  int wave = tid >> 5, lane = tid & 31;
  int lane15 = lane & 15, hi = lane >> 4, koff = hi * 8;

  {  // cooperative stage of weights + A' columns into LDS
    const uint4* src2 = (const uint4*)(W2H + net * 32768);
    uint4* dst2 = (uint4*)w2f;
    for (int idx = tid; idx < 4096; idx += 256) dst2[idx] = src2[idx];
    const uint4* src3 = (const uint4*)(W3H + net * 8192);
    uint4* dst3 = (uint4*)w3f;
    for (int idx = tid; idx < 1024; idx += 256) dst3[idx] = src3[idx];
    const uint4* srcA = (const uint4*)(Aph + ((size_t)(net * 2 + b) * 512 + ig * 8) * 256);
    uint4* dstA = (uint4*)acolh;
    if (tid < 256) dstA[tid] = srcA[tid];  // 8 rows x 256 halves = 4KB
    if (tid < 128) b2l[tid] = sb[net * 128 + tid];
    if (tid < 64) { b3l[tid] = sb[256 + net * 64 + tid]; w4l[tid] = sb[384 + net * 64 + tid]; }
    if (tid == 0) b4l[0] = sb[512 + net];
  }
  __syncthreads();

  int i = ig * 8 + wave;
  const _Float16* Bbase = Bph + (size_t)(net * 2 + b) * 512 * 256;
  const v8h* Av = (const v8h*)(acolh + wave * 256);
  float b4v = b4l[0];
  float* outp = net ? s2out : s1out;
  const v8h hzero = {};

  for (int it = 0; it < 8; ++it) {
    int jbase = jh * 256 + it * 32;

    // ---- conv1: h1 = relu(A'[c,i] + B'[c,j]) in packed f16 -> B-fragments
    v16h bf2[2][8];
    #pragma unroll
    for (int s = 0; s < 2; s++) {
      int j = jbase + s * 16 + lane15;
      const v8h* Bv = (const v8h*)(Bbase + (size_t)j * 256);
      #pragma unroll
      for (int kt = 0; kt < 8; kt++) {
        int c0 = kt * 32 + koff;  // multiple of 8
        v8h s0 = Av[c0 >> 3] + Bv[c0 >> 3];             // halves 0..7  (K=c0..c0+7)
        v8h s1 = Av[(c0 >> 3) + 2] + Bv[(c0 >> 3) + 2]; // halves 8..15 (K=c0+16..c0+23)
        s0 = __builtin_elementwise_max(s0, hzero);
        s1 = __builtin_elementwise_max(s1, hzero);
        bf2[s][kt] = __builtin_shufflevector(s0, s1, 0, 1, 2, 3, 4, 5, 6, 7,
                                             8, 9, 10, 11, 12, 13, 14, 15);
      }
    }

    // ---- prefetch next tile's B' columns into cache (global_prefetch_b8)
    if (it < 7) {
      int jn = jbase + 32 + lane15;
      const char* p0 = (const char*)(Bbase + (size_t)jn * 256);
      const char* p1 = (const char*)(Bbase + (size_t)(jn + 16) * 256);
      #pragma unroll
      for (int off = 0; off < 512; off += 128) {
        __builtin_prefetch(p0 + off, 0, 3);
        __builtin_prefetch(p1 + off, 0, 3);
      }
    }

    // ---- conv2: H2 = relu(W2' @ H1 + c2'); convert D-frags into conv3 B-frags
    v16h bf3[2][4];
    #pragma unroll
    for (int tp = 0; tp < 4; tp++) {
      v8f aA0 = {}, aA1 = {}, aB0 = {}, aB1 = {};   // inline C=0, no bias preload
      #pragma unroll
      for (int kt = 0; kt < 8; kt++) {
        v16h wA = *(const v16h*)(w2f + (size_t)(((2 * tp) * 8 + kt) * 32 + lane) * 16);
        v16h wB = *(const v16h*)(w2f + (size_t)(((2 * tp + 1) * 8 + kt) * 32 + lane) * 16);
        aA0 = wmma16(wA, bf2[0][kt], aA0);
        aA1 = wmma16(wA, bf2[1][kt], aA1);
        aB0 = wmma16(wB, bf2[0][kt], aB0);
        aB1 = wmma16(wB, bf2[1][kt], aB1);
      }
      // tail: bias + relu + f16 pack (short-lived bias loads, no spill pressure)
      float4 blo = *(const float4*)(b2l + 32 * tp + koff);
      float4 bhi = *(const float4*)(b2l + 32 * tp + koff + 4);
      float4 clo = *(const float4*)(b2l + 32 * tp + 16 + koff);
      float4 chi = *(const float4*)(b2l + 32 * tp + 16 + koff + 4);
      float bAv[8] = {blo.x, blo.y, blo.z, blo.w, bhi.x, bhi.y, bhi.z, bhi.w};
      float bBv[8] = {clo.x, clo.y, clo.z, clo.w, chi.x, chi.y, chi.z, chi.w};
      #pragma unroll
      for (int v = 0; v < 8; v++) {
        bf3[0][tp][v]     = (_Float16)fmaxf(aA0[v] + bAv[v], 0.f);
        bf3[0][tp][8 + v] = (_Float16)fmaxf(aB0[v] + bBv[v], 0.f);
        bf3[1][tp][v]     = (_Float16)fmaxf(aA1[v] + bAv[v], 0.f);
        bf3[1][tp][8 + v] = (_Float16)fmaxf(aB1[v] + bBv[v], 0.f);
      }
    }

    // ---- conv3: H3 = relu(W3' @ H2 + c3'); conv4: s = w4.H3 + b4
    float part0 = 0.f, part1 = 0.f;
    #pragma unroll
    for (int mt = 0; mt < 4; mt++) {
      v8f c30 = {}, c31 = {};
      #pragma unroll
      for (int kt = 0; kt < 4; kt++) {
        v16h w = *(const v16h*)(w3f + (size_t)((mt * 4 + kt) * 32 + lane) * 16);
        c30 = wmma16(w, bf3[0][kt], c30);
        c31 = wmma16(w, bf3[1][kt], c31);
      }
      float4 b3lo = *(const float4*)(b3l + 16 * mt + koff);
      float4 b3hi = *(const float4*)(b3l + 16 * mt + koff + 4);
      float4 wlo = *(const float4*)(w4l + 16 * mt + koff);
      float4 whi = *(const float4*)(w4l + 16 * mt + koff + 4);
      float b3v[8] = {b3lo.x, b3lo.y, b3lo.z, b3lo.w, b3hi.x, b3hi.y, b3hi.z, b3hi.w};
      float wv4[8] = {wlo.x, wlo.y, wlo.z, wlo.w, whi.x, whi.y, whi.z, whi.w};
      #pragma unroll
      for (int v = 0; v < 8; v++) {
        part0 += wv4[v] * fmaxf(c30[v] + b3v[v], 0.f);
        part1 += wv4[v] * fmaxf(c31[v] + b3v[v], 0.f);
      }
    }
    float t0 = part0 + __shfl_xor(part0, 16) + b4v;
    float t1 = part1 + __shfl_xor(part1, 16) + b4v;
    if (!hi) {
      size_t rb = ((size_t)b * 512 + i) * 512;
      outp[rb + jbase + lane15]      = t0;
      outp[rb + jbase + 16 + lane15] = t1;
    }
  }
}

// ---------------- K6: scores = s1 + s2^T ; init P = scores -------------------
__global__ __launch_bounds__(256) void k_combine(float* __restrict__ OUT,
                                                 const float* __restrict__ S2) {
  int row = blockIdx.x;  // b*512 + i
  int b = row >> 9, i = row & 511; int t = threadIdx.x;
  size_t rb = (size_t)row * 512;
  #pragma unroll
  for (int r = 0; r < 2; r++) {
    int j = t + 256 * r;
    float v = OUT[rb + j] + S2[((size_t)b * 512 + j) * 512 + i];
    OUT[rb + j] = v;
    OUT[524288 + rb + j] = v;  // P region
  }
}

// ---------------- Sinkhorn logsumexp normalizations --------------------------
__global__ __launch_bounds__(256) void k_lse_rows(float* __restrict__ P) {
  size_t rb = (size_t)blockIdx.x * 512;
  int t = threadIdx.x;
  __shared__ float red[256];
  float a = P[rb + t], c = P[rb + t + 256];
  red[t] = fmaxf(a, c); __syncthreads();
  #pragma unroll
  for (int s = 128; s > 0; s >>= 1) { if (t < s) red[t] = fmaxf(red[t], red[t + s]); __syncthreads(); }
  float m = red[0]; __syncthreads();
  red[t] = expf(a - m) + expf(c - m); __syncthreads();
  #pragma unroll
  for (int s = 128; s > 0; s >>= 1) { if (t < s) red[t] += red[t + s]; __syncthreads(); }
  float l = m + logf(red[0]);
  P[rb + t] = a - l; P[rb + t + 256] = c - l;
}

__global__ __launch_bounds__(256) void k_lse_cols(float* __restrict__ P) {
  int col = blockIdx.x; int b = col >> 9, j = col & 511; int t = threadIdx.x;
  size_t base = (size_t)b * 262144 + j;
  __shared__ float red[256];
  float a = P[base + (size_t)t * 512], c = P[base + (size_t)(t + 256) * 512];
  red[t] = fmaxf(a, c); __syncthreads();
  #pragma unroll
  for (int s = 128; s > 0; s >>= 1) { if (t < s) red[t] = fmaxf(red[t], red[t + s]); __syncthreads(); }
  float m = red[0]; __syncthreads();
  red[t] = expf(a - m) + expf(c - m); __syncthreads();
  #pragma unroll
  for (int s = 128; s > 0; s >>= 1) { if (t < s) red[t] += red[t + s]; __syncthreads(); }
  float l = m + logf(red[0]);
  P[base + (size_t)t * 512] = a - l; P[base + (size_t)(t + 256) * 512] = c - l;
}

// ---------------- host side --------------------------------------------------
// d_in flattening (dict insertion order, DFS over leaves):
//  0 image, 1 descriptors, 2 graph,
//  init: 3 l1.w 4 l1.b 5 n1.g 6 n1.b 7 l2.w 8 l2.b 9 n2.g 10 n2.b 11 l3.w 12 l3.b
//  tf layer L (base = 13+16L): wq.w/b wk.w/b wv.w/b wo.w/b ln1.g/b ln2.g/b ff1.w/b ff2.w/b
//  77 norm.g 78 norm.b
//  desc: 79 l1.w 80 l1.b 81 n1.g 82 n1.b 83 l2.w 84 l2.b 85 n2.g 86 n2.b
//  sn{1,2} (base = 87 + 20*net): c1.w/b b1.g/b/m/v c2.w/b b2.g/b/m/v c3.w/b b3.g/b/m/v c4.w/b
extern "C" void kernel_launch(void* const* d_in, const int* in_sizes, int n_in,
                              void* d_out, int out_size, void* d_ws, size_t ws_size,
                              hipStream_t stream) {
  (void)in_sizes; (void)n_in; (void)out_size; (void)ws_size;
  const float* desc = (const float*)d_in[1];
  const int* graph = (const int*)d_in[2];
#define PP(i) ((const float*)d_in[i])
  float* ws = (float*)d_ws;
  // f32 workspace (floats): X | Q | K | V | ATT (65536 each) | S2 (524288)
  float* X   = ws + 0;
  float* Q   = ws + 65536;
  float* Kb  = ws + 131072;
  float* V   = ws + 196608;
  float* ATT = ws + 262144;
  float* S2  = ws + 327680;
  _Float16* Aph = (_Float16*)((char*)d_ws + 3407872);  // 524288 halves
  _Float16* Bph = (_Float16*)((char*)d_ws + 4456448);  // 524288 halves
  _Float16* W2H = (_Float16*)((char*)d_ws + 5505024);  // 2 nets * 32768 halves
  _Float16* W3H = (_Float16*)((char*)d_ws + 5636096);  // 2 nets * 8192 halves
  float* sb     = (float*)((char*)d_ws + 5668864);     // folded biases / w4 / b4
  float* OUT = (float*)d_out;  // [0,524288): scores ; [524288,1048576): P

  // scorenet weight prep (BN fold + f16 fragment swizzle), both nets
  for (int net = 0; net < 2; ++net) {
    int s = 87 + net * 20;
    k_prep<<<32, 256, 0, stream>>>(
        PP(s + 6), PP(s + 7), PP(s + 8), PP(s + 9), PP(s + 10), PP(s + 11),
        PP(s + 12), PP(s + 13), PP(s + 14), PP(s + 15), PP(s + 16), PP(s + 17),
        PP(s + 18), PP(s + 19), W2H, W3H, sb, net);
  }

  k_gather_init<<<1024, 64, 0, stream>>>(
      desc, graph, PP(3), PP(4), PP(5), PP(6), PP(7), PP(8), PP(9), PP(10), PP(11), PP(12), X);

  for (int L = 0; L < 4; ++L) {
    int t0 = 13 + 16 * L;
    k_qkv<<<1024, 64, 0, stream>>>(X, PP(t0 + 0), PP(t0 + 1), PP(t0 + 2), PP(t0 + 3),
                                   PP(t0 + 4), PP(t0 + 5), Q, Kb, V);
    k_attn<<<16, 256, 0, stream>>>(Q, Kb, V, ATT);
    k_oproj<<<1024, 64, 0, stream>>>(X, ATT, PP(t0 + 6), PP(t0 + 7), PP(t0 + 8), PP(t0 + 9));
    k_ff<<<1024, 64, 0, stream>>>(X, PP(t0 + 12), PP(t0 + 13), PP(t0 + 14), PP(t0 + 15),
                                  PP(t0 + 10), PP(t0 + 11));
  }

  k_final<<<1024, 64, 0, stream>>>(X, PP(77), PP(78), PP(79), PP(80), PP(81), PP(82),
                                   PP(83), PP(84), PP(85), PP(86));

  for (int net = 0; net < 2; ++net) {
    int s = 87 + net * 20;
    k_ab<<<1024, 256, 0, stream>>>(X, PP(s + 0), PP(s + 1), PP(s + 2), PP(s + 3),
                                   PP(s + 4), PP(s + 5), Aph, Bph, net);
  }

  // dynamic LDS > 64KB: raise limit (idempotent; not a stream op, capture-safe)
  hipFuncSetAttribute(reinterpret_cast<const void*>(&k_score),
                      hipFuncAttributeMaxDynamicSharedMemorySize, SCORE_DYNLDS);
  k_score<<<512, 256, SCORE_DYNLDS, stream>>>(Aph, Bph, W2H, W3H, sb, OUT, S2);

  k_combine<<<1024, 256, 0, stream>>>(OUT, S2);

  float* Pm = OUT + 524288;
  for (int it = 0; it < 100; ++it) {
    k_lse_rows<<<1024, 256, 0, stream>>>(Pm);
    k_lse_cols<<<1024, 256, 0, stream>>>(Pm);
  }
#undef PP
}
